// TreeRNNCell_88210038325569
// MI455X (gfx1250) — matrix operand order repaired and checked by reading, not compiled, not run on hardware
//
#include <hip/hip_runtime.h>
#include <hip/hip_bf16.h>
#include <hip/hip_fp16.h>

// TreeRNN cell on MI455X (gfx1250): bf16 WMMA formulation.
//   out = tanh( (x*xmask) @ W_in  +  gathered_masked_h @ U_flat  + b_in*xmask + b_aggr )
// where U_flat is U[D,H,H] viewed as [512,64] and gathered_masked_h is [N,512].

typedef __attribute__((ext_vector_type(16))) __bf16 v16bf;
typedef __attribute__((ext_vector_type(8)))  float  v8f;

#define KX   128           // x_size
#define KU   512           // D*H
#define KTOT 640           // KX + KU
#define HH   64            // h_size

// ---------------------------------------------------------------------------
// Prep: build transposed bf16 B panel  Bt[col][k], col<64, k<640
//   k <  128 : W_in[k][col]
//   k >= 128 : U.flat[(k-128)*64 + col]
// ---------------------------------------------------------------------------
__global__ void prep_B_bf16(const float* __restrict__ W_in,
                            const float* __restrict__ U,
                            __bf16* __restrict__ Bt) {
    int idx = blockIdx.x * blockDim.x + threadIdx.x;
    if (idx >= HH * KTOT) return;
    int col = idx / KTOT;
    int k   = idx - col * KTOT;
    float v = (k < KX) ? W_in[k * HH + col] : U[(size_t)(k - KX) * HH + col];
    Bt[(size_t)col * KTOT + k] = (__bf16)v;
}

// ---------------------------------------------------------------------------
// Helpers
// ---------------------------------------------------------------------------
// Build an A fragment (16x32 bf16 layout) for this lane:
//   elements 0..7  = mem p0[0..7]  * s   (K = hi*16 + 0..7  of the 32-chunk)
//   elements 8..15 = mem p1[0..7]  * s   (K = hi*16 + 8..15 of the 32-chunk)
__device__ __forceinline__ v16bf make_a(const float* __restrict__ p0,
                                        const float* __restrict__ p1,
                                        float s) {
    float4 a0 = *(const float4*)(p0);
    float4 a1 = *(const float4*)(p0 + 4);
    float4 b0 = *(const float4*)(p1);
    float4 b1 = *(const float4*)(p1 + 4);
    v16bf r;
    r[0]  = (__bf16)(a0.x * s); r[1]  = (__bf16)(a0.y * s);
    r[2]  = (__bf16)(a0.z * s); r[3]  = (__bf16)(a0.w * s);
    r[4]  = (__bf16)(a1.x * s); r[5]  = (__bf16)(a1.y * s);
    r[6]  = (__bf16)(a1.z * s); r[7]  = (__bf16)(a1.w * s);
    r[8]  = (__bf16)(b0.x * s); r[9]  = (__bf16)(b0.y * s);
    r[10] = (__bf16)(b0.z * s); r[11] = (__bf16)(b0.w * s);
    r[12] = (__bf16)(b1.x * s); r[13] = (__bf16)(b1.y * s);
    r[14] = (__bf16)(b1.z * s); r[15] = (__bf16)(b1.w * s);
    return r;
}

// Load a B fragment (32x16 bf16 layout) = 16 contiguous bf16 from LDS
// (two ds_load_b128; avoid 32B-alignment assumptions on dynamic LDS base).
__device__ __forceinline__ v16bf ld_b(const __bf16* p) {
    union { uint4 q[2]; v16bf v; } u;
    u.q[0] = *(const uint4*)(p);
    u.q[1] = *(const uint4*)(p + 8);
    return u.v;
}

// ---------------------------------------------------------------------------
// Main kernel: 256 threads = 8 waves; each wave owns a stream of 16-node
// tiles and computes all 64 output columns (4 C tiles, 80 WMMAs per tile).
// B panel (80 KB bf16) is block-resident in LDS and re-read per tile via
// ds_load_b128 (LICM of those loads is defeated with an opaque offset so
// the compiler cannot hoist 640 VGPRs worth of B and spill it to scratch).
// ---------------------------------------------------------------------------
__global__ void __launch_bounds__(256, 1)
treernn_wmma_bf16(const float* __restrict__ xe,      // [N,128]
                  const float* __restrict__ xmask,   // [N]
                  const float* __restrict__ hprev,   // [N,64]
                  const float* __restrict__ cmask,   // [N,8]
                  const float* __restrict__ b_in,    // [64]
                  const float* __restrict__ b_aggr,  // [64]
                  const int*   __restrict__ cidx,    // [N,8]
                  const __bf16* __restrict__ Bt,     // [64,640] bf16 (ws)
                  float* __restrict__ out,           // [N,64]
                  int ntiles) {
    extern __shared__ __bf16 Blds[];                 // 64*640 bf16 = 80 KB

    const int tid = threadIdx.x;

    // Cooperative LDS fill of the B panel (16B chunks -> ds_store_b128).
    {
        const uint4* src = (const uint4*)Bt;
        uint4*       dst = (uint4*)Blds;
        const int nchunks = (HH * KTOT) / 8;         // 5120
        for (int i = tid; i < nchunks; i += blockDim.x) dst[i] = src[i];
    }
    __syncthreads();

    const int wave = tid >> 5;
    const int lane = tid & 31;
    const int row  = lane & 15;                      // A row / B col / C col idx
    const int hi   = lane >> 4;                      // half-wave selector

#pragma unroll 1
    for (int tile = blockIdx.x * 8 + wave; tile < ntiles; tile += gridDim.x * 8) {
        // Opaque zero offset, redefined every iteration: keeps all B-panel
        // ds_loads inside the loop (prevents LICM -> register spill storm).
        int obase = 0;
        asm volatile("" : "+v"(obase));

        const int   n    = tile * 16 + row;          // node this lane feeds into A
        const float xm   = xmask[n];
        const float* xrow = xe + (size_t)n * KX;

        // Prefetch next tile's A-side data (global_prefetch_b8).
        {
            int tile2 = tile + gridDim.x * 8;
            if (tile2 < ntiles) {
                int n2 = tile2 * 16 + row;
                __builtin_prefetch(xe + (size_t)n2 * KX, 0, 3);
                __builtin_prefetch(cidx + (size_t)n2 * 8, 0, 3);
            }
        }

        v8f acc[4];
#pragma unroll
        for (int t = 0; t < 4; ++t) {
            v8f z = {0.f, 0.f, 0.f, 0.f, 0.f, 0.f, 0.f, 0.f};
            acc[t] = z;
        }

        // ---- X part: (x * xmask) @ W_in ; K = 0..127, 4 steps of 32 ----
#pragma unroll
        for (int kk = 0; kk < 4; ++kk) {
            const int kb = kk * 32 + hi * 16;
            v16bf a = make_a(xrow + kb, xrow + kb + 8, xm);
            const int krow = kk * 32 + hi * 16;
#pragma unroll
            for (int t = 0; t < 4; ++t) {
                v16bf b = ld_b(&Blds[(size_t)((t * 16 + row) * KTOT + krow + obase)]);
                acc[t] = __builtin_amdgcn_wmma_f32_16x16x32_bf16(
                    false, a, false, b, (short)0, acc[t], false, false);
            }
        }

        // ---- Aggregation: gathered_masked_h @ U ; K = 128..639 ----
#pragma unroll
        for (int p = 0; p < 8; ++p) {
            const int   c = cidx[(size_t)n * 8 + p];
            const float m = cmask[(size_t)n * 8 + p];
            const float* hrow = hprev + (size_t)c * HH;
#pragma unroll
            for (int half = 0; half < 2; ++half) {       // two 32-chunks of the 64-wide child h
                const int kk = p * 2 + half;
                const int db = half * 32 + hi * 16;      // within-row offset in h (0..63)
                v16bf a = make_a(hrow + db, hrow + db + 8, m);
                const int krow = KX + kk * 32 + hi * 16;
#pragma unroll
                for (int t = 0; t < 4; ++t) {
                    v16bf b = ld_b(&Blds[(size_t)((t * 16 + row) * KTOT + krow + obase)]);
                    acc[t] = __builtin_amdgcn_wmma_f32_16x16x32_bf16(
                        false, a, false, b, (short)0, acc[t], false, false);
                }
            }
        }

        // ---- Epilogue: + b_in*xmask + b_aggr, tanh, store f32 ----
        float xm_ep[8];
#pragma unroll
        for (int r = 0; r < 8; ++r) xm_ep[r] = xmask[tile * 16 + hi * 8 + r];

#pragma unroll
        for (int t = 0; t < 4; ++t) {
            const int col = t * 16 + row;                // C/D: lane%16 = column
            const float bi = b_in[col];
            const float ba = b_aggr[col];
#pragma unroll
            for (int r = 0; r < 8; ++r) {
                const int node = tile * 16 + hi * 8 + r; // C/D: vgpr r -> row (+8 high half)
                out[(size_t)node * HH + col] = tanhf(acc[t][r] + bi * xm_ep[r] + ba);
            }
        }
    }
}

// ---------------------------------------------------------------------------
// Launch
// ---------------------------------------------------------------------------
extern "C" void kernel_launch(void* const* d_in, const int* in_sizes, int n_in,
                              void* d_out, int out_size, void* d_ws, size_t ws_size,
                              hipStream_t stream) {
    const float* xe     = (const float*)d_in[0];   // x_embs   [N,128]
    const float* xmask  = (const float*)d_in[1];   // x_mask   [N]
    const float* hprev  = (const float*)d_in[2];   // h_prev   [N,64]
    const float* cmask  = (const float*)d_in[3];   // child_mask [N,8]
    const float* W_in   = (const float*)d_in[4];   // [128,64]
    const float* b_in   = (const float*)d_in[5];   // [64]
    const float* U      = (const float*)d_in[6];   // [8,64,64]
    const float* b_aggr = (const float*)d_in[7];   // [64]
    const int*   cidx   = (const int*)d_in[8];     // child_idx [N,8]
    float*       out    = (float*)d_out;

    __bf16* Bt = (__bf16*)d_ws;                    // 64*640 bf16 = 80 KB

    const int N      = in_sizes[1];                // node count (x_mask size)
    const int ntiles = N / 16;

    {
        const int total = HH * KTOT;
        prep_B_bf16<<<(total + 255) / 256, 256, 0, stream>>>(W_in, U, Bt);
    }

    int blocks = 2048;                             // 8 waves/block; stride loop covers the rest
    if (blocks * 8 > ntiles) blocks = (ntiles + 7) / 8;
    if (blocks < 1) blocks = 1;

    const size_t ldsBytes = (size_t)HH * KTOT * 2; // 80 KB bf16 B panel
    treernn_wmma_bf16<<<blocks, 256, ldsBytes, stream>>>(
        xe, xmask, hprev, cmask, b_in, b_aggr, cidx, Bt, out, ntiles);
}
